// Multi_DefectModel_noFunc_22986664968795
// MI455X (gfx1250) — matrix operand order repaired and checked by reading, not compiled, not run on hardware
//
#include <hip/hip_runtime.h>
#include <hip/hip_bf16.h>
#include <hip/hip_fp16.h>

typedef __attribute__((ext_vector_type(16))) _Float16 v16h;
typedef __attribute__((ext_vector_type(8)))  _Float16 v8h;
typedef __attribute__((ext_vector_type(8)))  float    v8f;

#define BN_EPS 1e-5f
#define N_NODES 12800
#define N_B     128
#define N_PG    100
#define R_LD    112   // padded leading dim for affinity matrix

__device__ __forceinline__ float eluf(float x) { return x > 0.f ? x : (expf(x) - 1.f); }

// 16-byte global -> LDS copy: async on gfx1250 (ASYNCcnt-tracked), sync fallback.
// ISA: GLOBAL_LOAD_ASYNC_TO_LDS_B128  (VDST = per-lane LDS byte address,
// VADDR = 64-bit global address). LDS byte address = low 32 bits of generic ptr.
__device__ __forceinline__ void copy16(const _Float16* g, _Float16* l) {
#if defined(__gfx1250__)
    asm volatile("global_load_async_to_lds_b128 %0, %1, off"
                 :
                 : "v"((unsigned)(unsigned long long)l),
                   "v"((unsigned long long)g)
                 : "memory");
#else
    *(v8h*)l = *(const v8h*)g;
#endif
}
__device__ __forceinline__ void copy_wait() {
#if defined(__gfx1250__)
    asm volatile("s_wait_asynccnt 0x0" ::: "memory");
#endif
}

// A fragment: lane (m = lane&15, hh = lane>>4) reads As[row][hh*8 .. +7] and [16+hh*8 .. +7]
__device__ __forceinline__ v16h frag_a(const _Float16* row, int hh) {
    v8h lo = *(const v8h*)(row + hh * 8);
    v8h hi = *(const v8h*)(row + 16 + hh * 8);
    return __builtin_shufflevector(lo, hi, 0, 1, 2, 3, 4, 5, 6, 7, 8, 9, 10, 11, 12, 13, 14, 15);
}
// B fragment: lane (n = lane&15, hh) reads Bt[col][hh*8 .. +7] and [16+hh*8 .. +7] (n-major LDS tile)
__device__ __forceinline__ v16h frag_b(const _Float16* row, int hh) {
    v8h lo = *(const v8h*)(row + hh * 8);
    v8h hi = *(const v8h*)(row + 16 + hh * 8);
    return __builtin_shufflevector(lo, hi, 0, 1, 2, 3, 4, 5, 6, 7, 8, 9, 10, 11, 12, 13, 14, 15);
}

// ---------------------------------------------------------------------------
// elementwise staging helpers
// ---------------------------------------------------------------------------
__global__ void k_f32_to_f16(const float* __restrict__ s, _Float16* __restrict__ d, int n) {
    int i = blockIdx.x * 256 + threadIdx.x;
    if (i < n) d[i] = (_Float16)s[i];
}

// torch Linear weight [O,K] f32 -> [Npad,K] f16 (rows O..Npad zero)
__global__ void k_pad_rows(const float* __restrict__ w, _Float16* __restrict__ o,
                           int O, int K, int Np) {
    int i = blockIdx.x * 256 + threadIdx.x;
    if (i >= Np * K) return;
    int n = i / K;
    o[i] = (n < O) ? (_Float16)w[i] : (_Float16)0.f;
}

// GAT weight [K,N] f32 -> [N,K] f16 (transposed)
__global__ void k_transpose(const float* __restrict__ w, _Float16* __restrict__ o,
                            int K, int N) {
    int i = blockIdx.x * 256 + threadIdx.x;
    if (i >= N * K) return;
    int n = i / K, k = i % K;
    o[i] = (_Float16)w[(size_t)k * N + n];
}

__global__ void k_bn_lastdim_f16(const float* __restrict__ x, _Float16* __restrict__ o,
                                 const float* __restrict__ g, const float* __restrict__ b,
                                 const float* __restrict__ m, const float* __restrict__ v,
                                 int n, int C) {
    int i = blockIdx.x * 256 + threadIdx.x;
    if (i >= n) return;
    int c = i % C;
    o[i] = (_Float16)((x[i] - m[c]) * rsqrtf(v[c] + BN_EPS) * g[c] + b[c]);
}

// _bn_mid: params indexed by l = row % NPG
__global__ void k_bn_row_f16(const _Float16* __restrict__ x, _Float16* __restrict__ o,
                             const float* __restrict__ g, const float* __restrict__ b,
                             const float* __restrict__ m, const float* __restrict__ v,
                             int n, int C) {
    int i = blockIdx.x * 256 + threadIdx.x;
    if (i >= n) return;
    int l = (i / C) % N_PG;
    o[i] = (_Float16)(((float)x[i] - m[l]) * rsqrtf(v[l] + BN_EPS) * g[l] + b[l]);
}

// ---------------------------------------------------------------------------
// main WMMA GEMM: C[M,N] = A[M,K](f16, M%128==0) x W[Npad,K](f16, n-major)
// 128x64 block tile, 8 waves (4x2), 2x2 WMMA per wave, async double-buffered LDS
// epilogue: +bias[n], per-col BN affine, +res, ELU, dual f32/f16 store
// ---------------------------------------------------------------------------
__global__ __launch_bounds__(256)
void k_gemm(const _Float16* __restrict__ A, const _Float16* __restrict__ W,
            float* __restrict__ Cf, _Float16* __restrict__ Ch,
            const float* __restrict__ bias,
            const float* __restrict__ bng, const float* __restrict__ bnb,
            const float* __restrict__ bnm, const float* __restrict__ bnv,
            const float* __restrict__ res,
            int M, int N, int K, int ldc, int act) {
    __shared__ _Float16 As[2][128][40];
    __shared__ _Float16 Bt[2][64][40];
    const int tid  = threadIdx.x;
    const int lane = tid & 31, wid = tid >> 5;
    const int wm = wid >> 1, wn = wid & 1;
    const int m0 = blockIdx.y * 128, n0 = blockIdx.x * 64;
    const int mrow = lane & 15, hh = lane >> 4;
    const int cr = tid >> 2, cc = (tid & 3) * 8;   // copy row / k-chunk

    v8f acc[2][2];
    const v8f vz = {0.f, 0.f, 0.f, 0.f, 0.f, 0.f, 0.f, 0.f};
    acc[0][0] = vz; acc[0][1] = vz; acc[1][0] = vz; acc[1][1] = vz;

    const _Float16* Ag = A + (size_t)(m0 + cr) * K + cc;
    const _Float16* Wg = W + (size_t)(n0 + cr) * K + cc;

    // prologue: tile 0 into buffer 0
    copy16(Ag, &As[0][cr][cc]);
    copy16(Ag + (size_t)64 * K, &As[0][cr + 64][cc]);
    copy16(Wg, &Bt[0][cr][cc]);

    int p = 0;
    for (int k0 = 0; k0 < K; k0 += 32, p ^= 1) {
        copy_wait();
        __syncthreads();
        if (k0 + 32 < K) {   // overlap next tile fetch with WMMA
            copy16(Ag + k0 + 32, &As[p ^ 1][cr][cc]);
            copy16(Ag + (size_t)64 * K + k0 + 32, &As[p ^ 1][cr + 64][cc]);
            copy16(Wg + k0 + 32, &Bt[p ^ 1][cr][cc]);
        }
        v16h af[2], bf[2];
#pragma unroll
        for (int sm = 0; sm < 2; ++sm)
            af[sm] = frag_a(&As[p][wm * 32 + sm * 16 + mrow][0], hh);
#pragma unroll
        for (int sn = 0; sn < 2; ++sn)
            bf[sn] = frag_b(&Bt[p][wn * 32 + sn * 16 + mrow][0], hh);
#pragma unroll
        for (int sm = 0; sm < 2; ++sm)
#pragma unroll
            for (int sn = 0; sn < 2; ++sn)
                acc[sm][sn] = __builtin_amdgcn_wmma_f32_16x16x32_f16(
                    false, af[sm], false, bf[sn], (short)0, acc[sm][sn], false, false);
    }

#pragma unroll
    for (int sm = 0; sm < 2; ++sm)
#pragma unroll
        for (int sn = 0; sn < 2; ++sn)
#pragma unroll
            for (int v = 0; v < 8; ++v) {
                int m = m0 + wm * 32 + sm * 16 + hh * 8 + v;
                int n = n0 + wn * 32 + sn * 16 + mrow;
                if (n < N) {
                    float t = acc[sm][sn][v];
                    if (bias) t += bias[n];
                    if (bng)  t = (t - bnm[n]) * rsqrtf(bnv[n] + BN_EPS) * bng[n] + bnb[n];
                    if (res)  t += res[(size_t)m * ldc + n];
                    if (act == 1) t = eluf(t);
                    if (Cf) Cf[(size_t)m * ldc + n] = t;
                    if (Ch) Ch[(size_t)m * ldc + n] = (_Float16)t;
                }
            }
}

// ---------------------------------------------------------------------------
// batched affinity: R[b,n,m] = (1/NPG) * sum_i Th[b*100+n][i] * Ph[b*100+m][i]
// 64x64 tile, 4 waves, grid.z = batch; Ph is already [n][k] node-major
// ---------------------------------------------------------------------------
__global__ __launch_bounds__(128)
void k_rgemm(const _Float16* __restrict__ Th, const _Float16* __restrict__ Ph,
             _Float16* __restrict__ R) {
    const int b = blockIdx.z;
    const int tid = threadIdx.x, lane = tid & 31, wid = tid >> 5;
    const int wm = wid >> 1, wn = wid & 1;
    const int m0 = blockIdx.y * 64, n0 = blockIdx.x * 64;
    __shared__ _Float16 As[64][40];
    __shared__ _Float16 Bt[64][40];
    const _Float16* Tb = Th + (size_t)b * N_PG * 512;
    const _Float16* Pb = Ph + (size_t)b * N_PG * 512;

    v8f acc[2][2];
    const v8f vz = {0.f, 0.f, 0.f, 0.f, 0.f, 0.f, 0.f, 0.f};
    acc[0][0] = vz; acc[0][1] = vz; acc[1][0] = vz; acc[1][1] = vz;
    const v8h z8 = {};
    const int mrow = lane & 15, hh = lane >> 4;
    const int cr = tid >> 2, cc = (tid & 3) * 8;

    for (int k0 = 0; k0 < 512; k0 += 32) {
#pragma unroll
        for (int rr = 0; rr < 2; ++rr) {
            int r = cr + rr * 32;
            int gm = m0 + r, gn = n0 + r;
            *(v8h*)&As[r][cc] = (gm < N_PG) ? *(const v8h*)(Tb + (size_t)gm * 512 + k0 + cc) : z8;
            *(v8h*)&Bt[r][cc] = (gn < N_PG) ? *(const v8h*)(Pb + (size_t)gn * 512 + k0 + cc) : z8;
        }
        __syncthreads();
        v16h af[2], bf[2];
#pragma unroll
        for (int sm = 0; sm < 2; ++sm)
            af[sm] = frag_a(&As[wm * 32 + sm * 16 + mrow][0], hh);
#pragma unroll
        for (int sn = 0; sn < 2; ++sn)
            bf[sn] = frag_b(&Bt[wn * 32 + sn * 16 + mrow][0], hh);
#pragma unroll
        for (int sm = 0; sm < 2; ++sm)
#pragma unroll
            for (int sn = 0; sn < 2; ++sn)
                acc[sm][sn] = __builtin_amdgcn_wmma_f32_16x16x32_f16(
                    false, af[sm], false, bf[sn], (short)0, acc[sm][sn], false, false);
        __syncthreads();
    }
#pragma unroll
    for (int sm = 0; sm < 2; ++sm)
#pragma unroll
        for (int sn = 0; sn < 2; ++sn)
#pragma unroll
            for (int v = 0; v < 8; ++v) {
                int m = m0 + wm * 32 + sm * 16 + hh * 8 + v;
                int n = n0 + wn * 32 + sn * 16 + mrow;
                if (m < N_PG && n < N_PG)
                    R[(size_t)b * R_LD * R_LD + (size_t)m * R_LD + n] =
                        (_Float16)(acc[sm][sn][v] * (1.f / (float)N_PG));
            }
}

// ---------------------------------------------------------------------------
// batched y: Y[(b*100+n)*512 + i] = sum_m R[b][n][m] * G[(b*100+m)*512 + i]
// K = 100 zero-padded to 128; B tile transposed into n-major LDS
// ---------------------------------------------------------------------------
__global__ __launch_bounds__(128)
void k_ygemm(const _Float16* __restrict__ R, const _Float16* __restrict__ G,
             _Float16* __restrict__ Y) {
    const int b = blockIdx.z;
    const int tid = threadIdx.x, lane = tid & 31, wid = tid >> 5;
    const int wm = wid >> 1, wn = wid & 1;
    const int m0 = blockIdx.y * 64, n0 = blockIdx.x * 64;
    __shared__ _Float16 As[64][40];
    __shared__ _Float16 Bt[64][40];
    const _Float16* Rb = R + (size_t)b * R_LD * R_LD;
    const _Float16* Gb = G + (size_t)b * N_PG * 512;

    v8f acc[2][2];
    const v8f vz = {0.f, 0.f, 0.f, 0.f, 0.f, 0.f, 0.f, 0.f};
    acc[0][0] = vz; acc[0][1] = vz; acc[1][0] = vz; acc[1][1] = vz;
    const v8h z8 = {};
    const int mrow = lane & 15, hh = lane >> 4;
    const int cr = tid >> 2, cc = (tid & 3) * 8;

    for (int k0 = 0; k0 < 128; k0 += 32) {
        // A tile (R rows), chunk-guarded against ragged K=100
#pragma unroll
        for (int rr = 0; rr < 2; ++rr) {
            int r = cr + rr * 32;
            int gm = m0 + r, gk = k0 + cc;
            v8h val = z8;
            if (gm < N_PG) {
                if (gk + 8 <= N_PG) {
                    val = *(const v8h*)(Rb + (size_t)gm * R_LD + gk);
                } else {
#pragma unroll
                    for (int e = 0; e < 8; ++e)
                        val[e] = (gk + e < N_PG) ? Rb[(size_t)gm * R_LD + gk + e] : (_Float16)0.f;
                }
            }
            *(v8h*)&As[r][cc] = val;
        }
        // B tile: transpose G[k][n] -> Bt[n][k] (coalesced global reads)
        for (int i = tid; i < 32 * 64; i += 128) {
            int kk = i >> 6, c = i & 63;
            int gk = k0 + kk;
            Bt[c][kk] = (gk < N_PG) ? Gb[(size_t)gk * 512 + n0 + c] : (_Float16)0.f;
        }
        __syncthreads();
        v16h af[2], bf[2];
#pragma unroll
        for (int sm = 0; sm < 2; ++sm)
            af[sm] = frag_a(&As[wm * 32 + sm * 16 + mrow][0], hh);
#pragma unroll
        for (int sn = 0; sn < 2; ++sn)
            bf[sn] = frag_b(&Bt[wn * 32 + sn * 16 + mrow][0], hh);
#pragma unroll
        for (int sm = 0; sm < 2; ++sm)
#pragma unroll
            for (int sn = 0; sn < 2; ++sn)
                acc[sm][sn] = __builtin_amdgcn_wmma_f32_16x16x32_f16(
                    false, af[sm], false, bf[sn], (short)0, acc[sm][sn], false, false);
        __syncthreads();
    }
#pragma unroll
    for (int sm = 0; sm < 2; ++sm)
#pragma unroll
        for (int sn = 0; sn < 2; ++sn)
#pragma unroll
            for (int v = 0; v < 8; ++v) {
                int m = m0 + wm * 32 + sm * 16 + hh * 8 + v;
                int n = n0 + wn * 32 + sn * 16 + mrow;
                if (m < N_PG)
                    Y[((size_t)b * N_PG + m) * 512 + n] = (_Float16)acc[sm][sn][v];
            }
}

// ---------------------------------------------------------------------------
// GAT pieces
// ---------------------------------------------------------------------------
__global__ void k_elr(const _Float16* __restrict__ z, const float* __restrict__ al,
                      const float* __restrict__ ar, float* __restrict__ el,
                      float* __restrict__ er, int NH) {
    int i = blockIdx.x * 256 + threadIdx.x;
    if (i >= NH) return;
    int n = i >> 2, h = i & 3;
    const _Float16* zp = z + (size_t)n * 2048 + h * 512;
    const float* alp = al + h * 512;
    const float* arp = ar + h * 512;
    float sl = 0.f, sr = 0.f;
    for (int f = 0; f < 512; ++f) {
        float zv = (float)zp[f];
        sl += zv * alp[f];
        sr += zv * arp[f];
    }
    el[i] = sl; er[i] = sr;
}

__global__ void k_att(const float* __restrict__ el, const float* __restrict__ er,
                      const int* __restrict__ esrc, float* __restrict__ att, int N) {
    int i = blockIdx.x * 256 + threadIdx.x;
    if (i >= N * 4) return;
    int n = i >> 2, h = i & 3;
    float ei = er[i];
    float e[4];
    for (int j = 0; j < 4; ++j) {
        int s = esrc[n * 4 + j];
        float t = el[s * 4 + h] + ei;
        e[j] = t > 0.f ? t : 0.2f * t;
    }
    float mx = fmaxf(fmaxf(e[0], e[1]), fmaxf(e[2], e[3]));
    float a[4], ss = 0.f;
    for (int j = 0; j < 4; ++j) { a[j] = expf(e[j] - mx); ss += a[j]; }
    for (int j = 0; j < 4; ++j) att[n * 16 + h * 4 + j] = a[j] / ss;
}

__global__ void k_aggregate(const _Float16* __restrict__ z, const float* __restrict__ att,
                            const int* __restrict__ esrc, const float* __restrict__ bias,
                            _Float16* __restrict__ out, int N) {
    size_t i = (size_t)blockIdx.x * 256 + threadIdx.x;
    if (i >= (size_t)N * 2048) return;
    int n = (int)(i >> 11);
    int c = (int)(i & 2047);
    int h = c >> 9;
    float s = bias[c];
    for (int j = 0; j < 4; ++j) {
        int sc = esrc[n * 4 + j];
        s += att[n * 16 + h * 4 + j] * (float)z[(size_t)sc * 2048 + c];
    }
    out[i] = (_Float16)s;
}

// ---------------------------------------------------------------------------
// bbox path: bn_bbox (per-l) + fc_bbox(4->32) + elu -> v cols [480,512)
// ---------------------------------------------------------------------------
__global__ void k_bbox(const float* __restrict__ pos, const float* __restrict__ g,
                       const float* __restrict__ b, const float* __restrict__ m,
                       const float* __restrict__ v, const float* __restrict__ w,
                       const float* __restrict__ bias, float* __restrict__ vf,
                       _Float16* __restrict__ vh) {
    int i = blockIdx.x * 256 + threadIdx.x;
    if (i >= N_NODES * 32) return;
    int row = i >> 5, o = i & 31;
    int l = row % N_PG;
    float sc = rsqrtf(v[l] + BN_EPS) * g[l];
    float s = bias[o];
    for (int k = 0; k < 4; ++k) {
        float x = (pos[row * 4 + k] - m[l]) * sc + b[l];
        s += x * w[o * 4 + k];
    }
    s = eluf(s);
    vf[(size_t)row * 512 + 480 + o] = s;
    vh[(size_t)row * 512 + 480 + o] = (_Float16)s;
}

__global__ void k_normmean(const float* __restrict__ v, float* __restrict__ emd) {
    int i = blockIdx.x * 256 + threadIdx.x;
    if (i >= N_B * 512) return;
    int b = i >> 9, c = i & 511;
    const float* p = v + (size_t)b * N_PG * 512 + c;
    float s1 = 0.f, s2 = 0.f;
    for (int l = 0; l < N_PG; ++l) {
        float t = p[(size_t)l * 512];
        s1 += t; s2 += t * t;
    }
    emd[i] = s1 / ((float)N_PG * sqrtf(s2));
}

__global__ void k_final(const float* __restrict__ x, const float* __restrict__ emd,
                        const float* __restrict__ g, const float* __restrict__ bb,
                        const float* __restrict__ m, const float* __restrict__ v,
                        const float* __restrict__ w, const float* __restrict__ bias,
                        float* __restrict__ out) {
    int i = threadIdx.x;
    if (i >= 256) return;
    int b = i >> 1, o = i & 1;
    float s = bias[o];
    for (int c = 0; c < 1024; ++c) {
        float f = (c < 512) ? x[b * 512 + c] : emd[b * 512 + c - 512];
        f = (f - m[c]) * rsqrtf(v[c] + BN_EPS) * g[c] + bb[c];
        s += f * w[o * 1024 + c];
    }
    out[b * 2 + o] = s;
}

// ---------------------------------------------------------------------------
// host orchestration
// ---------------------------------------------------------------------------
namespace {
struct Ws {
    char* base; size_t off;
    void* take(size_t bytes) {
        off = (off + 255) & ~(size_t)255;
        void* p = base + off;
        off += bytes;
        return p;
    }
};

inline void gemm(hipStream_t s, const _Float16* A, const _Float16* W, float* Cf, _Float16* Ch,
                 const float* bias, const float* bng, const float* bnb, const float* bnm,
                 const float* bnv, const float* res, int M, int N, int K, int ldc, int act) {
    dim3 grid((N + 63) / 64, (M + 127) / 128);
    k_gemm<<<grid, 256, 0, s>>>(A, W, Cf, Ch, bias, bng, bnb, bnm, bnv, res,
                                M, N, K, ldc, act);
}
} // namespace

extern "C" void kernel_launch(void* const* d_in, const int* in_sizes, int n_in,
                              void* d_out, int out_size, void* d_ws, size_t ws_size,
                              hipStream_t stream) {
    (void)in_sizes; (void)n_in; (void)out_size; (void)ws_size;
    // ---- input index map (setup_inputs() dict insertion order, recursively) ----
    const float* IMG  = (const float*)d_in[0];
    const float* NODE = (const float*)d_in[2];
    const float* POS  = (const float*)d_in[3];
    const int*   ESRC = (const int*)d_in[4];
    const float* SWBN_G = (const float*)d_in[6],  *SWBN_B = (const float*)d_in[7];
    const float* SWBN_M = (const float*)d_in[8],  *SWBN_V = (const float*)d_in[9];
    const float* SWFC_W = (const float*)d_in[10], *SWFC_B = (const float*)d_in[11];
    const float* G1_W = (const float*)d_in[12], *G1_AL = (const float*)d_in[13];
    const float* G1_AR = (const float*)d_in[14], *G1_B = (const float*)d_in[15];
    const float* G2_W = (const float*)d_in[16], *G2_AL = (const float*)d_in[17];
    const float* G2_AR = (const float*)d_in[18], *G2_B = (const float*)d_in[19];
    const float* FC_W = (const float*)d_in[20], *FC_B = (const float*)d_in[21];
    const int HID = 22;                       // 8 x (w,b) -> 22..37
    const float* BG_G = (const float*)d_in[38], *BG_B = (const float*)d_in[39];
    const float* BG_M = (const float*)d_in[40], *BG_V = (const float*)d_in[41];
    const float* FG_W = (const float*)d_in[42], *FG_B = (const float*)d_in[43];
    const float* BB_G = (const float*)d_in[44], *BB_B = (const float*)d_in[45];
    const float* BB_M = (const float*)d_in[46], *BB_V = (const float*)d_in[47];
    const float* FB_W = (const float*)d_in[48], *FB_B = (const float*)d_in[49];
    const int RS = 50;                        // 8 x 12 -> 50..145
    const float* FN_G = (const float*)d_in[146], *FN_B = (const float*)d_in[147];
    const float* FN_M = (const float*)d_in[148], *FN_V = (const float*)d_in[149];
    const float* FF_W = (const float*)d_in[150], *FF_B = (const float*)d_in[151];

    Ws ws{(char*)d_ws, 0};
    // f16 weight staging, all stored [Npad][K] n-major
    _Float16* w_swin  = (_Float16*)ws.take((size_t)512 * 1024 * 2);
    _Float16* w_g1    = (_Float16*)ws.take((size_t)2048 * 768 * 2);
    _Float16* w_g2    = (_Float16*)ws.take((size_t)2048 * 2048 * 2);
    _Float16* w_fc    = (_Float16*)ws.take((size_t)512 * 2048 * 2);
    _Float16* w_hid[8];
    for (int i = 0; i < 8; ++i) w_hid[i] = (_Float16*)ws.take((size_t)512 * 512 * 2);
    _Float16* w_fcgat = (_Float16*)ws.take((size_t)512 * 512 * 2);
    _Float16* w_rs[8][4];
    for (int t = 0; t < 8; ++t)
        for (int q = 0; q < 4; ++q) w_rs[t][q] = (_Float16*)ws.take((size_t)512 * 512 * 2);
    // activations
    _Float16* imgbn = (_Float16*)ws.take((size_t)N_B * 1024 * 2);
    float*    x512  = (float*)ws.take((size_t)N_B * 512 * 4);
    _Float16* nodeh = (_Float16*)ws.take((size_t)N_NODES * 768 * 2);
    _Float16* bufA  = (_Float16*)ws.take((size_t)N_NODES * 2048 * 2);
    _Float16* bufB  = (_Float16*)ws.take((size_t)N_NODES * 2048 * 2);
    float*    el    = (float*)ws.take((size_t)N_NODES * 4 * 4);
    float*    er    = (float*)ws.take((size_t)N_NODES * 4 * 4);
    float*    att   = (float*)ws.take((size_t)N_NODES * 16 * 4);
    _Float16* hP    = (_Float16*)ws.take((size_t)N_NODES * 512 * 2);
    _Float16* hQ    = (_Float16*)ws.take((size_t)N_NODES * 512 * 2);
    float*    vf32  = (float*)ws.take((size_t)N_NODES * 512 * 4);
    _Float16* vh    = (_Float16*)ws.take((size_t)N_NODES * 512 * 2);
    _Float16* gbuf  = (_Float16*)ws.take((size_t)N_NODES * 512 * 2);
    _Float16* thbuf = (_Float16*)ws.take((size_t)N_NODES * 512 * 2);
    _Float16* phbuf = (_Float16*)ws.take((size_t)N_NODES * 512 * 2);
    _Float16* ybuf  = (_Float16*)ws.take((size_t)N_NODES * 512 * 2);
    _Float16* Rbuf  = (_Float16*)ws.take((size_t)N_B * R_LD * R_LD * 2);
    float*    emd   = (float*)ws.take((size_t)N_B * 512 * 4);

    auto cdiv = [](long a, long b) { return (unsigned)((a + b - 1) / b); };

    // ---- stage weights ----
    k_pad_rows<<<cdiv(512L * 1024, 256), 256, 0, stream>>>(SWFC_W, w_swin, 512, 1024, 512);
    k_transpose<<<cdiv(2048L * 768, 256), 256, 0, stream>>>(G1_W, w_g1, 768, 2048);
    k_transpose<<<cdiv(2048L * 2048, 256), 256, 0, stream>>>(G2_W, w_g2, 2048, 2048);
    k_pad_rows<<<cdiv(512L * 2048, 256), 256, 0, stream>>>(FC_W, w_fc, 512, 2048, 512);
    for (int i = 0; i < 8; ++i)
        k_pad_rows<<<cdiv(512L * 512, 256), 256, 0, stream>>>(
            (const float*)d_in[HID + 2 * i], w_hid[i], 512, 512, 512);
    k_pad_rows<<<cdiv(512L * 512, 256), 256, 0, stream>>>(FG_W, w_fcgat, 480, 512, 512);
    for (int t = 0; t < 8; ++t) {
        const int base = RS + t * 12;
        k_pad_rows<<<cdiv(512L * 512, 256), 256, 0, stream>>>(
            (const float*)d_in[base + 0], w_rs[t][0], 512, 512, 512);   // g
        k_pad_rows<<<cdiv(512L * 512, 256), 256, 0, stream>>>(
            (const float*)d_in[base + 2], w_rs[t][1], 512, 512, 512);   // theta
        k_pad_rows<<<cdiv(512L * 512, 256), 256, 0, stream>>>(
            (const float*)d_in[base + 4], w_rs[t][2], 512, 512, 512);   // phi
        k_pad_rows<<<cdiv(512L * 512, 256), 256, 0, stream>>>(
            (const float*)d_in[base + 6], w_rs[t][3], 512, 512, 512);   // w
    }

    // ---- swin path ----
    k_bn_lastdim_f16<<<cdiv((long)N_B * 1024, 256), 256, 0, stream>>>(
        IMG, imgbn, SWBN_G, SWBN_B, SWBN_M, SWBN_V, N_B * 1024, 1024);
    gemm(stream, imgbn, w_swin, x512, nullptr, SWFC_B, nullptr, nullptr, nullptr, nullptr,
         nullptr, N_B, 512, 1024, 512, 1);

    // ---- GAT1 ----
    k_f32_to_f16<<<cdiv((long)N_NODES * 768, 256), 256, 0, stream>>>(NODE, nodeh, N_NODES * 768);
    gemm(stream, nodeh, w_g1, nullptr, bufA, nullptr, nullptr, nullptr, nullptr, nullptr,
         nullptr, N_NODES, 2048, 768, 2048, 0);
    k_elr<<<cdiv((long)N_NODES * 4, 256), 256, 0, stream>>>(bufA, G1_AL, G1_AR, el, er, N_NODES * 4);
    k_att<<<cdiv((long)N_NODES * 4, 256), 256, 0, stream>>>(el, er, ESRC, att, N_NODES);
    k_aggregate<<<cdiv((long)N_NODES * 2048, 256), 256, 0, stream>>>(bufA, att, ESRC, G1_B, bufB, N_NODES);

    // ---- GAT2 ----
    gemm(stream, bufB, w_g2, nullptr, bufA, nullptr, nullptr, nullptr, nullptr, nullptr,
         nullptr, N_NODES, 2048, 2048, 2048, 0);
    k_elr<<<cdiv((long)N_NODES * 4, 256), 256, 0, stream>>>(bufA, G2_AL, G2_AR, el, er, N_NODES * 4);
    k_att<<<cdiv((long)N_NODES * 4, 256), 256, 0, stream>>>(el, er, ESRC, att, N_NODES);
    k_aggregate<<<cdiv((long)N_NODES * 2048, 256), 256, 0, stream>>>(bufA, att, ESRC, G2_B, bufB, N_NODES);

    // ---- fc + 8 hidden ----
    gemm(stream, bufB, w_fc, nullptr, hP, FC_B, nullptr, nullptr, nullptr, nullptr, nullptr,
         N_NODES, 512, 2048, 512, 1);
    _Float16 *cur = hP, *nxt = hQ;
    for (int i = 0; i < 8; ++i) {
        gemm(stream, cur, w_hid[i], nullptr, nxt, (const float*)d_in[HID + 2 * i + 1],
             nullptr, nullptr, nullptr, nullptr, nullptr, N_NODES, 512, 512, 512, 1);
        _Float16* t = cur; cur = nxt; nxt = t;
    }

    // ---- bn_gat + fc_gat(512->480) -> v[:,0:480); bbox -> v[:,480:512) ----
    k_bn_row_f16<<<cdiv((long)N_NODES * 512, 256), 256, 0, stream>>>(
        cur, nxt, BG_G, BG_B, BG_M, BG_V, N_NODES * 512, 512);
    gemm(stream, nxt, w_fcgat, vf32, vh, FG_B, nullptr, nullptr, nullptr, nullptr, nullptr,
         N_NODES, 480, 512, 512, 1);
    k_bbox<<<cdiv((long)N_NODES * 32, 256), 256, 0, stream>>>(
        POS, BB_G, BB_B, BB_M, BB_V, FB_W, FB_B, vf32, vh);

    // ---- 8 x rs_gcn non-local blocks ----
    for (int t = 0; t < 8; ++t) {
        const int base = RS + t * 12;
        gemm(stream, vh, w_rs[t][0], nullptr, gbuf, (const float*)d_in[base + 1],
             nullptr, nullptr, nullptr, nullptr, nullptr, N_NODES, 512, 512, 512, 0);
        gemm(stream, vh, w_rs[t][1], nullptr, thbuf, (const float*)d_in[base + 3],
             nullptr, nullptr, nullptr, nullptr, nullptr, N_NODES, 512, 512, 512, 0);
        gemm(stream, vh, w_rs[t][2], nullptr, phbuf, (const float*)d_in[base + 5],
             nullptr, nullptr, nullptr, nullptr, nullptr, N_NODES, 512, 512, 512, 0);
        k_rgemm<<<dim3(2, 2, N_B), 128, 0, stream>>>(thbuf, phbuf, Rbuf);
        k_ygemm<<<dim3(8, 2, N_B), 128, 0, stream>>>(Rbuf, gbuf, ybuf);
        gemm(stream, ybuf, w_rs[t][3], vf32, vh, (const float*)d_in[base + 7],
             (const float*)d_in[base + 8], (const float*)d_in[base + 9],
             (const float*)d_in[base + 10], (const float*)d_in[base + 11],
             vf32, N_NODES, 512, 512, 512, 0);
    }

    // ---- l2norm over nodes + mean, final bn + fc ----
    k_normmean<<<cdiv((long)N_B * 512, 256), 256, 0, stream>>>(vf32, emd);
    k_final<<<1, 256, 0, stream>>>(x512, emd, FN_G, FN_B, FN_M, FN_V, FF_W, FF_B, (float*)d_out);
}
